// ModalEncoder_80985903333774
// MI455X (gfx1250) — compile-verified, matrix operand order checked by reading, not run on hardware
//
#include <hip/hip_runtime.h>
#include <hip/hip_bf16.h>
#include <math.h>

// ---------------------------------------------------------------------------
// MI455X (gfx1250, wave32) fused ModalEncoder.
// Attention is algebraically folded (q is length-1 and per-batch constant),
// so the FLOP-dominant work is the FFN, executed as f16 WMMA GEMMs
// (v_wmma_f32_16x16x32_f16) with f32 accumulation.
// Round 3 -> Round 4: k_ffn moves to M_BLK=32 with 2x2 tile grids per wave so
// every A/B fragment feeds 2 WMMAs (round 3 had 1 WMMA per B fragment and a
// full load->wait0->wmma stall per tile). A tile staged in LDS (16 KB), act
// tile in LDS (64 KB); 80 KB total still allows multiple blocks per WGP.
// ---------------------------------------------------------------------------

typedef __attribute__((ext_vector_type(16))) _Float16 v16h;
typedef __attribute__((ext_vector_type(8)))  _Float16 v8h;
typedef __attribute__((ext_vector_type(8)))  float    v8f;

#define BN   4
#define NN   1024
#define FF   32
#define II   8
#define DIM  256
#define DN   64
#define HH   4
#define HD   64
#define EPSF 1e-5f
#define MBLK 32

// -------------------- helpers --------------------

__device__ __forceinline__ float block_sum256(float v, float* rbuf, int t) {
  rbuf[t] = v; __syncthreads();
  #pragma unroll
  for (int s = 128; s > 0; s >>= 1) {
    if (t < s) rbuf[t] += rbuf[t + s];
    __syncthreads();
  }
  float r = rbuf[0]; __syncthreads();
  return r;
}

__device__ __forceinline__ float gelu_tanh(float x) {
  float x3 = x * x * x;
  return 0.5f * x * (1.0f + tanhf(0.7978845608028654f * (x + 0.044715f * x3)));
}

// Load one 16x32 f16 fragment (A) or 32x16 fragment (B, stored n-major) from a
// row-major f16 buffer with leading dimension `ld`.
// ISA 7.12.2 layout: lanes 0-15 hold K 0..7 / 16..23, lanes 16-31 hold
// K 8..15 / 24..31, row/col = lane & 15. Two contiguous 16B loads per lane.
__device__ __forceinline__ v16h load_frag16(const _Float16* src, int ld, int lane) {
  int m  = lane & 15;
  int kb = (lane < 16) ? 0 : 8;
  const _Float16* p = src + (size_t)m * ld + kb;
  v8h lo = *reinterpret_cast<const v8h*>(p);
  v8h hi = *reinterpret_cast<const v8h*>(p + 16);
  v16h r;
  #pragma unroll
  for (int i = 0; i < 8; ++i) { r[i] = lo[i]; r[i + 8] = hi[i]; }
  return r;
}

// -------------------- kernel 1: per-batch precompute --------------------
// s/sh vectors for the three cLNs, plus folded query projection
// wkq[b][h][d] = sum_e Wk[d, h*64+e] * qh[b, h*64+e].
__global__ void k_batch(const float* __restrict__ ctx, const float* __restrict__ q_emb,
                        const float* __restrict__ ckvW, const float* __restrict__ ckvB,
                        const float* __restrict__ cqW,  const float* __restrict__ cqB,
                        const float* __restrict__ chW,  const float* __restrict__ chB,
                        const float* __restrict__ Wq,   const float* __restrict__ Wk,
                        float* __restrict__ sskv, float* __restrict__ sshv,
                        float* __restrict__ wkq) {
  __shared__ float cs[DN];
  __shared__ float qn_s[DIM];
  __shared__ float qh_s[DIM];
  __shared__ float rbuf[256];
  int t = threadIdx.x, b = blockIdx.x;
  if (t < DN) cs[t] = ctx[b * DN + t];
  __syncthreads();

  float s0 = 0, s1 = 0, s2 = 0, s3 = 0, s4 = 0, s5 = 0;
  for (int j = 0; j < DN; ++j) {
    float c = cs[j];
    s0 += c * ckvW[j * (2 * DIM) + t];
    s1 += c * ckvW[j * (2 * DIM) + DIM + t];
    s2 += c * cqW [j * (2 * DIM) + t];
    s3 += c * cqW [j * (2 * DIM) + DIM + t];
    s4 += c * chW [j * (2 * DIM) + t];
    s5 += c * chW [j * (2 * DIM) + DIM + t];
  }
  s0 += ckvB[t]; s1 += ckvB[DIM + t];
  s2 += cqB[t];  s3 += cqB[DIM + t];
  s4 += chB[t];  s5 += chB[DIM + t];
  sskv[b * 2 * DIM + t]       = s0;
  sskv[b * 2 * DIM + DIM + t] = s1;
  sshv[b * 2 * DIM + t]       = s4;
  sshv[b * 2 * DIM + DIM + t] = s5;

  // qn = cln(q_emb): LN over 256, then per-batch scale/shift
  float q   = q_emb[t];
  float mq  = block_sum256(q, rbuf, t) * (1.0f / DIM);
  float qd  = q - mq;
  float vq  = block_sum256(qd * qd, rbuf, t) * (1.0f / DIM);
  float qn  = qd * rsqrtf(vq + EPSF);
  qn = qn * (1.0f + s2) + s3;
  qn_s[t] = qn; __syncthreads();

  // qh = qn @ Wq
  float qh = 0;
  for (int e = 0; e < DIM; ++e) qh += qn_s[e] * Wq[e * DIM + t];
  qh_s[t] = qh; __syncthreads();

  // fold Wk into the query: wkq[h][d]
  for (int h = 0; h < HH; ++h) {
    float a = 0;
    for (int e = 0; e < HD; ++e) a += Wk[t * DIM + h * HD + e] * qh_s[h * HD + e];
    wkq[b * HH * DIM + h * DIM + t] = a;
  }
}

// -------------------- kernel 2: f32 -> f16 weight transposes --------------------
// W1t[n][k] (1024 x 256) and W2t[n][k] (256 x 1024), n-major so WMMA B
// fragments are contiguous 16B loads.
__global__ void k_convert(const float* __restrict__ W1, const float* __restrict__ W2,
                          _Float16* __restrict__ W1t, _Float16* __restrict__ W2t) {
  int i = blockIdx.x * blockDim.x + threadIdx.x;  // 0 .. 262143
  { int n = i >> 8,  k = i & 255;  W1t[i] = (_Float16)W1[k * (4 * DIM) + n]; }
  { int n = i >> 10, k = i & 1023; W2t[i] = (_Float16)W2[k * DIM + n]; }
}

// -------------------- kernel 3: fused per-token attention --------------------
__global__ void k_token(const float* __restrict__ x, const int* __restrict__ idx,
                        const float* __restrict__ Wp, const float* __restrict__ Bp,
                        const float* __restrict__ q_emb,
                        const float* __restrict__ Wv, const float* __restrict__ Wo,
                        const float* __restrict__ sskv, const float* __restrict__ sshv,
                        const float* __restrict__ wkq,
                        float* __restrict__ hbuf, _Float16* __restrict__ hnbuf) {
  __shared__ float xs[FF * II];
  __shared__ float kvs[FF * DIM];     // 32 KB
  __shared__ float rp[256], rp2[256];
  __shared__ float m_s[FF], r_s[FF];
  __shared__ float att_s[HH * FF];
  __shared__ float smax[HH], ssum[HH];
  __shared__ float u_s[HH * DIM];
  __shared__ float o_s[DIM];

  int t = threadIdx.x;
  int token = blockIdx.x;
  int b = token >> 10;

  xs[t] = x[(size_t)token * (FF * II) + t];
  __syncthreads();

  float skv = sskv[b * 2 * DIM + t], shkv = sskv[b * 2 * DIM + DIM + t];

  // kv[f][d=t] = sum_i x[f,i] * Wp[idx[f]][d*8+i]
  for (int f = 0; f < FF; ++f) {
    int rid = idx[f];
    const float* wrow = Wp + (size_t)rid * (DIM * II) + t * II;
    float acc = 0;
    #pragma unroll
    for (int i = 0; i < II; ++i) acc += xs[f * II + i] * wrow[i];
    kvs[f * DIM + t] = acc;
  }
  __syncthreads();

  // per-row LN stats (32 rows, 8 threads per row)
  {
    int g = t >> 3, l8 = t & 7;
    float s = 0, s2 = 0;
    for (int d = l8 * 32; d < l8 * 32 + 32; ++d) {
      float v = kvs[g * DIM + d]; s += v; s2 += v * v;
    }
    rp[t] = s; rp2[t] = s2;
  }
  __syncthreads();
  if (t < FF) {
    float s = 0, s2 = 0;
    #pragma unroll
    for (int j = 0; j < 8; ++j) { s += rp[t * 8 + j]; s2 += rp2[t * 8 + j]; }
    float m   = s  * (1.0f / DIM);
    float var = s2 * (1.0f / DIM) - m * m;
    m_s[t] = m; r_s[t] = rsqrtf(var + EPSF);
  }
  __syncthreads();

  // normalize + per-batch cLN + gathered bias
  for (int f = 0; f < FF; ++f) {
    float v = (kvs[f * DIM + t] - m_s[f]) * r_s[f];
    v = v * (1.0f + skv) + shkv + Bp[(size_t)idx[f] * DIM + t];
    kvs[f * DIM + t] = v;
  }
  __syncthreads();

  // attention logits via folded wkq: l[h,f] = (kv[f,:] . wkq[h,:]) / 8
  if (t < HH * FF) {
    int h = t >> 5, f = t & 31;
    const float* wq = wkq + b * HH * DIM + h * DIM;
    float l = 0;
    for (int d = 0; d < DIM; ++d) l += kvs[f * DIM + d] * wq[d];
    att_s[t] = l * 0.125f;   // 1/sqrt(64)
  }
  __syncthreads();
  if (t < HH) {
    float mx = -1e30f;
    for (int f = 0; f < FF; ++f) mx = fmaxf(mx, att_s[t * FF + f]);
    smax[t] = mx;
  }
  __syncthreads();
  if (t < HH * FF) att_s[t] = __expf(att_s[t] - smax[t >> 5]);
  __syncthreads();
  if (t < HH) {
    float s = 0;
    for (int f = 0; f < FF; ++f) s += att_s[t * FF + f];
    ssum[t] = s;
  }
  __syncthreads();
  if (t < HH * FF) att_s[t] = att_s[t] / ssum[t >> 5];
  __syncthreads();

  // u[h][d] = sum_f att[h,f] * kv[f,d]
  #pragma unroll
  for (int h = 0; h < HH; ++h) {
    float u = 0;
    for (int f = 0; f < FF; ++f) u += att_s[h * FF + f] * kvs[f * DIM + t];
    u_s[h * DIM + t] = u;
  }
  __syncthreads();

  // o[h*64+e] = sum_d u[h,d] * Wv[d, h*64+e]   (t = h*64+e)
  {
    int h = t >> 6;
    const float* uh = u_s + h * DIM;
    float o = 0;
    for (int d = 0; d < DIM; ++d) o += uh[d] * Wv[d * DIM + t];
    o_s[t] = o;
  }
  __syncthreads();

  // h = q_emb + o @ Wo ; hn = cln_h(h)
  float wo = 0;
  for (int d = 0; d < DIM; ++d) wo += o_s[d] * Wo[d * DIM + t];
  float hv  = q_emb[t] + wo;
  float mh  = block_sum256(hv, rp, t) * (1.0f / DIM);
  float hd  = hv - mh;
  float vh  = block_sum256(hd * hd, rp, t) * (1.0f / DIM);
  float hn  = hd * rsqrtf(vh + EPSF);
  hn = hn * (1.0f + sshv[b * 2 * DIM + t]) + sshv[b * 2 * DIM + DIM + t];

  hbuf [(size_t)token * DIM + t] = hv;
  hnbuf[(size_t)token * DIM + t] = (_Float16)hn;
}

// -------------------- kernel 4: WMMA FFN (the FLOP-dominant part) ------------
// Per block: 32 token-rows, 8 wave32s (256 threads), grid = 128 blocks.
// GEMM1: a_s(32x256, LDS) x W1t -> GELU -> act LDS (32x1024 f16, 64 KB).
//        4 N-passes; per pass each wave computes a 2x2 grid of 16x16 tiles:
//        each A/B fragment feeds 2 WMMAs (vs 1 in round 3).
// GEMM2: act(LDS) x W2t -> + residual + b2 -> out; same 2x2 tiling.
__global__ void __launch_bounds__(256, 1)
k_ffn(const _Float16* __restrict__ hn,
      const _Float16* __restrict__ W1t, const float* __restrict__ bias1,
      const _Float16* __restrict__ W2t, const float* __restrict__ bias2,
      const float* __restrict__ hbuf, float* __restrict__ out) {
  __shared__ __align__(16) _Float16 a_s[MBLK * DIM];    // 16 KB staged hn tile
  __shared__ __align__(16) _Float16 act[MBLK * 1024];   // 64 KB activations
  int t = threadIdx.x;
  int lane = t & 31, wave = t >> 5;
  int row0  = blockIdx.x * MBLK;
  int col15 = lane & 15;
  int half8 = (lane >> 4) << 3;   // rows j or j+8 per C/D layout

  // cooperative stage of the hn tile: 32 rows x 256 cols f16 = 16 KB contiguous
  {
    const uint4* src = (const uint4*)(hn + (size_t)row0 * DIM);
    uint4* dst = (uint4*)a_s;
    #pragma unroll
    for (int i = t; i < (MBLK * DIM) / 8; i += 256) dst[i] = src[i];
  }
  __syncthreads();

  // ---- GEMM1: 4 passes over N=1024; each wave owns 32 cols per pass ----
  for (int pass = 0; pass < 4; ++pass) {
    int n0 = pass * 256 + wave * 32;
    v8f acc[2][2];
    #pragma unroll
    for (int mt = 0; mt < 2; ++mt)
      #pragma unroll
      for (int nt = 0; nt < 2; ++nt) {
        float bv = bias1[n0 + nt * 16 + col15];
        #pragma unroll
        for (int j = 0; j < 8; ++j) acc[mt][nt][j] = bv;
      }
    for (int k0 = 0; k0 < DIM; k0 += 32) {
      v16h bf0 = load_frag16(W1t + (size_t)n0 * DIM + k0, DIM, lane);
      v16h bf1 = load_frag16(W1t + (size_t)(n0 + 16) * DIM + k0, DIM, lane);
      v16h a0  = load_frag16(a_s + k0, DIM, lane);             // ds_load_b128
      v16h a1  = load_frag16(a_s + 16 * DIM + k0, DIM, lane);
      acc[0][0] = __builtin_amdgcn_wmma_f32_16x16x32_f16(false, a0, false, bf0, (short)0, acc[0][0], false, false);
      acc[0][1] = __builtin_amdgcn_wmma_f32_16x16x32_f16(false, a0, false, bf1, (short)0, acc[0][1], false, false);
      acc[1][0] = __builtin_amdgcn_wmma_f32_16x16x32_f16(false, a1, false, bf0, (short)0, acc[1][0], false, false);
      acc[1][1] = __builtin_amdgcn_wmma_f32_16x16x32_f16(false, a1, false, bf1, (short)0, acc[1][1], false, false);
    }
    #pragma unroll
    for (int mt = 0; mt < 2; ++mt)
      #pragma unroll
      for (int nt = 0; nt < 2; ++nt) {
        int col = n0 + nt * 16 + col15;
        #pragma unroll
        for (int j = 0; j < 8; ++j) {
          int r = mt * 16 + half8 + j;
          act[r * 1024 + col] = (_Float16)gelu_tanh(acc[mt][nt][j]);
        }
      }
  }
  __syncthreads();

  // ---- GEMM2: each wave owns 32 cols of the 256 outputs ----
  {
    int n0 = wave * 32;
    v8f acc[2][2];
    #pragma unroll
    for (int mt = 0; mt < 2; ++mt)
      #pragma unroll
      for (int nt = 0; nt < 2; ++nt) {
        float bv = bias2[n0 + nt * 16 + col15];
        #pragma unroll
        for (int j = 0; j < 8; ++j) acc[mt][nt][j] = bv;
      }
    for (int k0 = 0; k0 < 1024; k0 += 32) {
      v16h bf0 = load_frag16(W2t + (size_t)n0 * 1024 + k0, 1024, lane);
      v16h bf1 = load_frag16(W2t + (size_t)(n0 + 16) * 1024 + k0, 1024, lane);
      v16h a0  = load_frag16(act + k0, 1024, lane);            // ds_load_b128
      v16h a1  = load_frag16(act + 16 * 1024 + k0, 1024, lane);
      acc[0][0] = __builtin_amdgcn_wmma_f32_16x16x32_f16(false, a0, false, bf0, (short)0, acc[0][0], false, false);
      acc[0][1] = __builtin_amdgcn_wmma_f32_16x16x32_f16(false, a0, false, bf1, (short)0, acc[0][1], false, false);
      acc[1][0] = __builtin_amdgcn_wmma_f32_16x16x32_f16(false, a1, false, bf0, (short)0, acc[1][0], false, false);
      acc[1][1] = __builtin_amdgcn_wmma_f32_16x16x32_f16(false, a1, false, bf1, (short)0, acc[1][1], false, false);
    }
    #pragma unroll
    for (int mt = 0; mt < 2; ++mt)
      #pragma unroll
      for (int nt = 0; nt < 2; ++nt) {
        int col = n0 + nt * 16 + col15;
        #pragma unroll
        for (int j = 0; j < 8; ++j) {
          int grow = row0 + mt * 16 + half8 + j;
          out[(size_t)grow * DIM + col] = hbuf[(size_t)grow * DIM + col] + acc[mt][nt][j];
        }
      }
  }
}

// -------------------- host launcher --------------------
extern "C" void kernel_launch(void* const* d_in, const int* in_sizes, int n_in,
                              void* d_out, int out_size, void* d_ws, size_t ws_size,
                              hipStream_t stream) {
  (void)in_sizes; (void)n_in; (void)out_size; (void)ws_size;
  const float* x     = (const float*)d_in[0];
  const int*   idx   = (const int*)  d_in[1];
  const float* ctx   = (const float*)d_in[2];
  const float* Wp    = (const float*)d_in[3];
  const float* Bp    = (const float*)d_in[4];
  const float* q_emb = (const float*)d_in[5];
  const float* ckvW  = (const float*)d_in[6];
  const float* ckvB  = (const float*)d_in[7];
  const float* cqW   = (const float*)d_in[8];
  const float* cqB   = (const float*)d_in[9];
  const float* chW   = (const float*)d_in[10];
  const float* chB   = (const float*)d_in[11];
  const float* Wq    = (const float*)d_in[12];
  const float* Wk    = (const float*)d_in[13];
  const float* Wv    = (const float*)d_in[14];
  const float* Wo    = (const float*)d_in[15];
  const float* W1    = (const float*)d_in[16];
  const float* b1    = (const float*)d_in[17];
  const float* W2    = (const float*)d_in[18];
  const float* b2    = (const float*)d_in[19];

  // workspace layout (all offsets 16B-aligned), total ~7.04 MB
  char* ws = (char*)d_ws;
  float*    sskv  = (float*)   (ws + 0);        // 4*512 f32
  float*    sshv  = (float*)   (ws + 8192);     // 4*512 f32
  float*    wkq   = (float*)   (ws + 16384);    // 4*1024 f32
  _Float16* W1t   = (_Float16*)(ws + 32768);    // 1024*256 f16
  _Float16* W2t   = (_Float16*)(ws + 557056);   // 256*1024 f16
  float*    hbuf  = (float*)   (ws + 1081344);  // 4096*256 f32
  _Float16* hnbuf = (_Float16*)(ws + 5275648);  // 4096*256 f16

  float* outp = (float*)d_out;

  k_batch  <<<BN,   256, 0, stream>>>(ctx, q_emb, ckvW, ckvB, cqW, cqB, chW, chB,
                                      Wq, Wk, sskv, sshv, wkq);
  k_convert<<<1024, 256, 0, stream>>>(W1, W2, W1t, W2t);
  k_token  <<<BN * NN, 256, 0, stream>>>(x, idx, Wp, Bp, q_emb, Wv, Wo,
                                         sskv, sshv, wkq, hbuf, hnbuf);
  k_ffn    <<<(BN * NN) / MBLK, 256, 0, stream>>>(hnbuf, W1t, b1, W2t, b2, hbuf, outp);
}
